// GPT_91018946936940
// MI455X (gfx1250) — compile-verified, matrix operand order checked by reading, not run on hardware
//
#include <hip/hip_runtime.h>
#include <math.h>

// Model dims (fixed by reference)
#define BDIM   1024
#define NH     16
#define DKH    64
#define SEQ    512
#define BATCH  4
#define DFF    4096
#define NLAYER 6
#define VOCAB  32000
#define MTOT   (BATCH*SEQ)   // 2048 rows

typedef __attribute__((ext_vector_type(16))) __bf16 v16bf;
typedef __attribute__((ext_vector_type(8)))  __bf16 v8bf;
typedef __attribute__((ext_vector_type(8)))  float  v8f;

typedef int v4i32 __attribute__((vector_size(16)));
typedef __attribute__((address_space(1))) v4i32* as1_v4i32p;
typedef __attribute__((address_space(3))) v4i32* as3_v4i32p;

#define ASYNC_OK __has_builtin(__builtin_amdgcn_global_load_async_to_lds_b128)

__device__ __forceinline__ unsigned short f2bf(float f) {
  unsigned int u = __float_as_uint(f);
  u += 0x7FFFu + ((u >> 16) & 1u);   // round-to-nearest-even
  return (unsigned short)(u >> 16);
}

// Copy 16B global -> LDS. gfx1250 async path (ASYNCcnt) with sync fallback.
__device__ __forceinline__ void cp16_g2l(const void* g, void* l) {
#if ASYNC_OK
  __builtin_amdgcn_global_load_async_to_lds_b128(
      (as1_v4i32p)(uintptr_t)g, (as3_v4i32p)(uintptr_t)l, 0, 0);
#else
  *(uint4*)l = *(const uint4*)g;
#endif
}
__device__ __forceinline__ void wait_async0() {
#if ASYNC_OK
# if __has_builtin(__builtin_amdgcn_s_wait_asynccnt)
  __builtin_amdgcn_s_wait_asynccnt(0);
# else
  asm volatile("s_wait_asynccnt 0x0" ::: "memory");
# endif
#endif
}

// Build a 16x32 bf16 WMMA operand fragment from an LDS row.
// Lane layout (ISA 7.12.2): lanes 0-15 -> K=0..7 (v0..3), K=16..23 (v4..7);
// lanes 16-31 -> K=8..15, K=24..31.  => two contiguous 16B LDS loads per lane.
__device__ __forceinline__ v16bf frag_from_lds(const unsigned short* rowbase, int g) {
  const v8bf lo = *(const v8bf*)(rowbase + g * 8);
  const v8bf hi = *(const v8bf*)(rowbase + 16 + g * 8);
  v16bf r;
#pragma unroll
  for (int i = 0; i < 8; ++i) { r[i] = lo[i]; r[i + 8] = hi[i]; }
  return r;
}

// C[M,N] = act( A[M,K] @ Wt[N,K]^T + bias ) (+ res)
// A, Wt are bf16; 256 thr (8 waves), 128x128 tile, K-step 32, double-buffered
// async global->LDS copies overlapped with 8 v_wmma_f32_16x16x32_bf16 per step.
template<bool BIAS, bool RES, bool GELU_ACT, bool OUT_BF16>
__global__ __launch_bounds__(256) void gemm_wmma(
    const unsigned short* __restrict__ A,   // [M,K] bf16
    const unsigned short* __restrict__ Wt,  // [N,K] bf16 (pre-transposed)
    const float* __restrict__ bias, const float* __restrict__ res,
    void* __restrict__ Cout, int M, int N, int K)
{
  __shared__ __align__(16) unsigned short As[2][128 * 40];  // [row][k], pad->40
  __shared__ __align__(16) unsigned short Bs[2][128 * 40];  // [n][k]

  const int tid  = threadIdx.x;
  const int wave = tid >> 5;
  const int lane = tid & 31;
  const int lg   = lane >> 4;       // half-wave group
  const int ln16 = lane & 15;
  const int wm   = wave >> 1;       // 0..3 (M)
  const int wn   = wave & 1;        // 0..1 (N)
  const int bm   = blockIdx.y * 128;
  const int bn   = blockIdx.x * 128;

  v8f acc[2][4];
#pragma unroll
  for (int i = 0; i < 2; ++i)
#pragma unroll
    for (int j = 0; j < 4; ++j)
#pragma unroll
      for (int e = 0; e < 8; ++e) acc[i][j][e] = 0.0f;

  const int nk = K >> 5;

  // Each tile = 128 rows x 32 bf16 = 512 x 16B chunks; 2 chunks/thread/tile.
  auto issue_tiles = [&](int kt, int buf) {
    const int k0 = kt << 5;
#pragma unroll
    for (int cc = 0; cc < 2; ++cc) {
      const int c   = tid + cc * 256;
      const int row = c >> 2;
      const int ko  = (c & 3) * 8;
      cp16_g2l(A  + (size_t)(bm + row) * K + k0 + ko, &As[buf][row * 40 + ko]);
      cp16_g2l(Wt + (size_t)(bn + row) * K + k0 + ko, &Bs[buf][row * 40 + ko]);
    }
  };

  issue_tiles(0, 0);
  for (int kt = 0; kt < nk; ++kt) {
    const int buf = kt & 1;
    wait_async0();           // this wave's copies for tile kt are done
    __syncthreads();         // all waves' copies visible; prev-buf readers done
    if (kt + 1 < nk) issue_tiles(kt + 1, buf ^ 1);  // overlap with WMMAs below

    v16bf af[2], bf[4];
#pragma unroll
    for (int tm = 0; tm < 2; ++tm)
      af[tm] = frag_from_lds(&As[buf][(wm * 32 + tm * 16 + ln16) * 40], lg);
#pragma unroll
    for (int tn = 0; tn < 4; ++tn)
      bf[tn] = frag_from_lds(&Bs[buf][(wn * 64 + tn * 16 + ln16) * 40], lg);

#pragma unroll
    for (int tm = 0; tm < 2; ++tm)
#pragma unroll
      for (int tn = 0; tn < 4; ++tn)
        acc[tm][tn] = __builtin_amdgcn_wmma_f32_16x16x32_bf16(
            false, af[tm], false, bf[tn], (short)0, acc[tm][tn], false, false);
  }

  // Epilogue: C/D layout -> lane 0-15: N=lane, M=r; lane 16-31: N=lane-16, M=r+8
  float* Cf = (float*)Cout;
  unsigned short* Cb = (unsigned short*)Cout;
#pragma unroll
  for (int tm = 0; tm < 2; ++tm) {
#pragma unroll
    for (int tn = 0; tn < 4; ++tn) {
      const int n = bn + wn * 64 + tn * 16 + ln16;
#pragma unroll
      for (int r = 0; r < 8; ++r) {
        const int m = bm + wm * 32 + tm * 16 + r + 8 * lg;
        float vv = acc[tm][tn][r];
        if (BIAS) vv += bias[n];
        if (GELU_ACT) vv = 0.5f * vv * (1.0f + erff(vv * 0.70710678118f));
        const size_t off = (size_t)m * N + n;
        if (RES) vv += res[off];
        if (OUT_BF16) Cb[off] = f2bf(vv); else Cf[off] = vv;
      }
    }
  }
}

// W[K,N] f32 -> Wt[N,K] bf16, 32x32 LDS-tiled transpose (both sides coalesced)
__global__ __launch_bounds__(256) void transpose_bf_k(const float* __restrict__ W,
    unsigned short* __restrict__ Wt, int K, int N)
{
  __shared__ unsigned short tile[32][33];
  const int tx = threadIdx.x & 31;
  const int ty = threadIdx.x >> 5;       // 0..7
  const int k0 = blockIdx.y * 32;
  const int n0 = blockIdx.x * 32;
#pragma unroll
  for (int i = 0; i < 4; ++i) {
    const int k = ty + 8 * i;
    tile[k][tx] = f2bf(W[(size_t)(k0 + k) * N + n0 + tx]);
  }
  __syncthreads();
#pragma unroll
  for (int i = 0; i < 4; ++i) {
    const int n = ty + 8 * i;
    Wt[(size_t)(n0 + n) * K + k0 + tx] = tile[tx][n];
  }
}

__global__ __launch_bounds__(256) void embed_k(const int* __restrict__ x,
    const float* __restrict__ emb, float* __restrict__ h)
{
  const int row = blockIdx.x;            // b*SEQ + s
  const int s   = row % SEQ;
  const int tok = x[row];
  const float* er = emb + (size_t)tok * BDIM;
  float* hr = h + (size_t)row * BDIM;
  const float pos = (float)s;
  const int t = threadIdx.x;
#pragma unroll
  for (int i = 0; i < 4; ++i) {
    const int d  = t + 256 * i;
    const int i2 = d & ~1;
    const float dv  = __expf((float)i2 * (-9.2103403719761836f / (float)BDIM));
    const float ang = pos * dv;
    const float pe  = (d & 1) ? __cosf(ang) : __sinf(ang);
    hr[d] = er[d] + pe;
  }
}

// LayerNorm, f32 in -> bf16 out (feeds GEMM A operands directly)
__global__ __launch_bounds__(256) void layernorm_bf_k(const float* __restrict__ x,
    const float* __restrict__ g, const float* __restrict__ b,
    unsigned short* __restrict__ y)
{
  __shared__ float red[256];
  const int row = blockIdx.x, t = threadIdx.x;
  const float* xr = x + (size_t)row * BDIM;
  float v[4]; float s = 0.f;
#pragma unroll
  for (int i = 0; i < 4; ++i) { v[i] = xr[t + 256 * i]; s += v[i]; }
  red[t] = s; __syncthreads();
  for (int o = 128; o > 0; o >>= 1) { if (t < o) red[t] += red[t + o]; __syncthreads(); }
  const float mu = red[0] * (1.0f / BDIM); __syncthreads();
  float qv = 0.f;
#pragma unroll
  for (int i = 0; i < 4; ++i) { const float d = v[i] - mu; qv += d * d; }
  red[t] = qv; __syncthreads();
  for (int o = 128; o > 0; o >>= 1) { if (t < o) red[t] += red[t + o]; __syncthreads(); }
  const float rstd = rsqrtf(red[0] * (1.0f / BDIM) + 1e-5f);
  unsigned short* yr = y + (size_t)row * BDIM;
#pragma unroll
  for (int i = 0; i < 4; ++i) {
    const int d = t + 256 * i;
    yr[d] = f2bf((v[i] - mu) * rstd * g[d] + b[d]);
  }
}

// One block (64 thr, 2 waves) per (b, head, query). Two-pass causal softmax.
// q,k,v f32; output ctx bf16 (feeds o-proj GEMM).
__global__ __launch_bounds__(64) void attn_k(const float* __restrict__ q,
    const float* __restrict__ k, const float* __restrict__ v,
    unsigned short* __restrict__ out)
{
  __shared__ float sc[SEQ];
  __shared__ float red[64];
  const int t = threadIdx.x;
  int idx = blockIdx.x;
  const int qi = idx % SEQ; idx /= SEQ;
  const int hh = idx % NH;  idx /= NH;
  const int bb = idx;
  const float* qp = q + (size_t)(bb * SEQ + qi) * BDIM + hh * DKH;

  float lmax = -1e30f;
  for (int j = t; j <= qi; j += 64) {
    const float* kp = k + (size_t)(bb * SEQ + j) * BDIM + hh * DKH;
    float s = 0.f;
#pragma unroll 8
    for (int d = 0; d < DKH; ++d) s += qp[d] * kp[d];
    s *= 0.125f;                 // 1/sqrt(64)
    sc[j] = s;
    lmax = fmaxf(lmax, s);
  }
  red[t] = lmax; __syncthreads();
  for (int o = 32; o > 0; o >>= 1) { if (t < o) red[t] = fmaxf(red[t], red[t + o]); __syncthreads(); }
  const float mx = red[0]; __syncthreads();

  float lsum = 0.f;
  for (int j = t; j <= qi; j += 64) { const float e = __expf(sc[j] - mx); sc[j] = e; lsum += e; }
  red[t] = lsum; __syncthreads();
  for (int o = 32; o > 0; o >>= 1) { if (t < o) red[t] += red[t + o]; __syncthreads(); }
  const float inv = 1.0f / red[0];
  __syncthreads();

  float acc = 0.f;
  const float* vbase = v + (size_t)(bb * SEQ) * BDIM + hh * DKH + t;  // t = head dim
  for (int j = 0; j <= qi; ++j) acc += sc[j] * vbase[(size_t)j * BDIM];
  out[(size_t)(bb * SEQ + qi) * BDIM + hh * DKH + t] = f2bf(acc * inv);
}

extern "C" void kernel_launch(void* const* d_in, const int* in_sizes, int n_in,
                              void* d_out, int out_size, void* d_ws, size_t ws_size,
                              hipStream_t stream)
{
  (void)in_sizes; (void)n_in; (void)out_size; (void)ws_size;
  const int*   x    = (const int*)  d_in[0];
  const float* temb = (const float*)d_in[1];
  const float* wq   = (const float*)d_in[2];
  const float* bq   = (const float*)d_in[3];
  const float* wk   = (const float*)d_in[4];
  const float* bk   = (const float*)d_in[5];
  const float* wv   = (const float*)d_in[6];
  const float* bv   = (const float*)d_in[7];
  const float* wo   = (const float*)d_in[8];
  const float* bo   = (const float*)d_in[9];
  const float* ln1g = (const float*)d_in[10];
  const float* ln1b = (const float*)d_in[11];
  const float* w1   = (const float*)d_in[12];
  const float* b1   = (const float*)d_in[13];
  const float* w2   = (const float*)d_in[14];
  const float* b2   = (const float*)d_in[15];
  const float* ln2g = (const float*)d_in[16];
  const float* ln2b = (const float*)d_in[17];
  const float* lnfg = (const float*)d_in[18];
  const float* lnfb = (const float*)d_in[19];
  const float* lmw  = (const float*)d_in[20];

  const size_t SZ = (size_t)MTOT * BDIM;       // 2,097,152
  float* h  = (float*)d_ws;
  float* qb = h  + SZ;
  float* kb = qb + SZ;
  float* vb = kb + SZ;
  unsigned short* ab0 = (unsigned short*)(vb + SZ);   // 8M bf16 (activations A)
  unsigned short* ab1 = ab0 + (size_t)MTOT * DFF;     // 8M bf16 (ctx / gelu out)
  unsigned short* wt  = ab1 + (size_t)MTOT * DFF;     // 32.8M bf16 (Wt scratch)

  const dim3 blk(256);
  const dim3 gD(BDIM  / 128, MTOT / 128);  //   8 x 16
  const dim3 gF(DFF   / 128, MTOT / 128);  //  32 x 16
  const dim3 gV(VOCAB / 128, MTOT / 128);  // 250 x 16
  const dim3 tDD(BDIM / 32, BDIM / 32);    // D x D transpose
  const dim3 tDF(DFF / 32, BDIM / 32);     // w1: K=D, N=DFF
  const dim3 tFD(BDIM / 32, DFF / 32);     // w2: K=DFF, N=D
  const dim3 tDV(VOCAB / 32, BDIM / 32);   // lm_w

  embed_k<<<MTOT, blk, 0, stream>>>(x, temb, h);

  for (int l = 0; l < NLAYER; ++l) {
    const size_t wofs = (size_t)l * BDIM * BDIM;
    const size_t bofs = (size_t)l * BDIM;
    const size_t w1o  = (size_t)l * BDIM * DFF;
    const size_t b1o  = (size_t)l * DFF;

    layernorm_bf_k<<<MTOT, blk, 0, stream>>>(h, ln1g + bofs, ln1b + bofs, ab0);

    transpose_bf_k<<<tDD, blk, 0, stream>>>(wq + wofs, wt, BDIM, BDIM);
    gemm_wmma<true,false,false,false><<<gD, blk, 0, stream>>>(ab0, wt, bq + bofs, nullptr, qb, MTOT, BDIM, BDIM);
    transpose_bf_k<<<tDD, blk, 0, stream>>>(wk + wofs, wt, BDIM, BDIM);
    gemm_wmma<true,false,false,false><<<gD, blk, 0, stream>>>(ab0, wt, bk + bofs, nullptr, kb, MTOT, BDIM, BDIM);
    transpose_bf_k<<<tDD, blk, 0, stream>>>(wv + wofs, wt, BDIM, BDIM);
    gemm_wmma<true,false,false,false><<<gD, blk, 0, stream>>>(ab0, wt, bv + bofs, nullptr, vb, MTOT, BDIM, BDIM);

    attn_k<<<BATCH * NH * SEQ, dim3(64), 0, stream>>>(qb, kb, vb, ab1);

    transpose_bf_k<<<tDD, blk, 0, stream>>>(wo + wofs, wt, BDIM, BDIM);
    gemm_wmma<true,true,false,false><<<gD, blk, 0, stream>>>(ab1, wt, bo + bofs, h, h, MTOT, BDIM, BDIM);

    layernorm_bf_k<<<MTOT, blk, 0, stream>>>(h, ln2g + bofs, ln2b + bofs, ab0);

    transpose_bf_k<<<tDF, blk, 0, stream>>>(w1 + w1o, wt, BDIM, DFF);
    gemm_wmma<true,false,true,true><<<gF, blk, 0, stream>>>(ab0, wt, b1 + b1o, nullptr, ab1, MTOT, DFF, BDIM);
    transpose_bf_k<<<tFD, blk, 0, stream>>>(w2 + w1o, wt, DFF, BDIM);
    gemm_wmma<true,true,false,false><<<gD, blk, 0, stream>>>(ab1, wt, b2 + bofs, h, h, MTOT, BDIM, DFF);
  }

  layernorm_bf_k<<<MTOT, blk, 0, stream>>>(h, lnfg, lnfb, ab0);
  transpose_bf_k<<<tDV, blk, 0, stream>>>(lmw, wt, BDIM, VOCAB);
  gemm_wmma<false,false,false,false><<<gV, blk, 0, stream>>>(ab0, wt, nullptr, nullptr, (float*)d_out, MTOT, VOCAB, BDIM);
}
